// CrossCAM_Module_69389491634870
// MI455X (gfx1250) — matrix operand order repaired and checked by reading, not compiled, not run on hardware
//
#include <hip/hip_runtime.h>
#include <hip/hip_bf16.h>
#include <cstddef>

typedef __attribute__((ext_vector_type(16))) __bf16 v16bf;
typedef __attribute__((ext_vector_type(8)))  __bf16 v8bf;
typedef __attribute__((ext_vector_type(4)))  __bf16 v4bf;
typedef __attribute__((ext_vector_type(8)))  float  v8f;

#define B_   16
#define C_   64
#define N_   16384      // H*W = 128*128
#define KSPLIT 8
#define KC   128        // K chunk staged in LDS per iteration (energy kernel)
#define PAD  8          // LDS row padding in bf16 elements (16 bytes)

static __device__ inline v16bf cat8(v8bf lo, v8bf hi) {
  v16bf r;
#pragma unroll
  for (int i = 0; i < 8; ++i) { r[i] = lo[i]; r[i + 8] = hi[i]; }
  return r;
}

static __device__ inline v8bf pack8(float4 a, float4 b) {
  v8bf p;
  p[0] = (__bf16)a.x; p[1] = (__bf16)a.y; p[2] = (__bf16)a.z; p[3] = (__bf16)a.w;
  p[4] = (__bf16)b.x; p[5] = (__bf16)b.y; p[6] = (__bf16)b.z; p[7] = (__bf16)b.w;
  return p;
}

static __device__ inline v8f wmma_bf16(v16bf a, v16bf b, v8f c) {
  // v_wmma_f32_16x16x32_bf16 : D = A(16x32) * B(32x16) + C, f32 accumulate
  return __builtin_amdgcn_wmma_f32_16x16x32_bf16(
      /*neg_a=*/false, a, /*neg_b=*/false, b,
      /*c_mod=*/(short)0, c, /*reuse_a=*/false, /*reuse_b=*/false);
}

// ---------------------------------------------------------------------------
// Kernel 1: energy[b][m] = V V^T   (V = map[b] viewed as [64][16384]), bf16
// WMMA with f32 accumulation, K split 8 ways across blocks, atomic reduce.
// grid = B*2*KSPLIT blocks of 256 threads (8 waves, 2 16x16 tiles per wave,
// both tiles in the same row so the A fragment is shared/CSE'd).
// ---------------------------------------------------------------------------
__global__ __launch_bounds__(256) void energy_kernel(
    const float* __restrict__ tmap, const float* __restrict__ rmap,
    float* __restrict__ energy /* [B][2][64][64] */) {
  __shared__ __attribute__((aligned(16))) __bf16 sV[C_][KC + PAD];

  const int blk = blockIdx.x;
  const int ks  = blk % KSPLIT;
  const int bm  = blk / KSPLIT;          // (b*2 + m)
  const int m   = bm & 1;
  const int b   = bm >> 1;
  const float* __restrict__ src =
      (m == 0 ? tmap : rmap) + (size_t)b * C_ * N_;

  const int t    = threadIdx.x;
  const int lane = t & 31;
  const int wave = t >> 5;               // 0..7
  const int lm   = lane & 15;
  const bool hiH = lane >= 16;

  // tile assignment: 4x4 grid of 16x16 tiles; wave owns tiles (ci, cj0/cj0+1)
  const int t0  = wave * 2;
  const int ci0 = t0 >> 2, cj0 = t0 & 3;
  const int ci1 = (t0 + 1) >> 2, cj1 = (t0 + 1) & 3;   // ci1 == ci0

  v8f acc0 = {}; v8f acc1 = {};

  const int ldRow = t >> 2;              // 0..63
  const int ldK   = (t & 3) * 32;        // 0,32,64,96

  const int kBeg = ks * (N_ / KSPLIT);
  const int kEnd = kBeg + (N_ / KSPLIT);

  for (int kb = kBeg; kb < kEnd; kb += KC) {
    // ---- stage 64x128 f32 chunk -> bf16 LDS (each thread: 32 floats) ----
    const float* gp = src + (size_t)ldRow * N_ + kb + ldK;
    float4 f[8];
#pragma unroll
    for (int i = 0; i < 8; ++i) f[i] = ((const float4*)gp)[i];
    v8bf p0 = pack8(f[0], f[1]);
    v8bf p1 = pack8(f[2], f[3]);
    v8bf p2 = pack8(f[4], f[5]);
    v8bf p3 = pack8(f[6], f[7]);
    __syncthreads();                     // previous iteration's reads done
    *(v8bf*)&sV[ldRow][ldK]      = p0;
    *(v8bf*)&sV[ldRow][ldK + 8]  = p1;
    *(v8bf*)&sV[ldRow][ldK + 16] = p2;
    *(v8bf*)&sV[ldRow][ldK + 24] = p3;
    __syncthreads();                     // fills visible

    // ---- four 32-wide K steps ----
#pragma unroll
    for (int kk = 0; kk < KC; kk += 32) {
      const __bf16* ar = &sV[ci0 * 16 + lm][kk];
      v16bf a = cat8(*(const v8bf*)(ar + (hiH ? 16 : 0)),
                     *(const v8bf*)(ar + (hiH ? 24 : 8)));
      const __bf16* br0 = &sV[cj0 * 16 + lm][kk + (hiH ? 16 : 0)];
      v16bf bb0 = cat8(*(const v8bf*)br0, *(const v8bf*)(br0 + 8));
      acc0 = wmma_bf16(a, bb0, acc0);
      const __bf16* br1 = &sV[cj1 * 16 + lm][kk + (hiH ? 16 : 0)];
      v16bf bb1 = cat8(*(const v8bf*)br1, *(const v8bf*)(br1 + 8));
      acc1 = wmma_bf16(a, bb1, acc1);
    }
  }

  // ---- reduce partial tiles into energy via f32 global atomics ----
  float* eb = energy + (size_t)bm * 64 * 64;
#pragma unroll
  for (int v = 0; v < 8; ++v) {
    int r0 = ci0 * 16 + v + (hiH ? 8 : 0);
    __hip_atomic_fetch_add(&eb[r0 * 64 + cj0 * 16 + lm], acc0[v],
                           __ATOMIC_RELAXED, __HIP_MEMORY_SCOPE_AGENT);
    int r1 = ci1 * 16 + v + (hiH ? 8 : 0);
    __hip_atomic_fetch_add(&eb[r1 * 64 + cj1 * 16 + lm], acc1[v],
                           __ATOMIC_RELAXED, __HIP_MEMORY_SCOPE_AGENT);
  }
}

// ---------------------------------------------------------------------------
// Kernel 2: softmax(rowmax(e)-e) per row  ->  M[b] = [Wt(g*Ar+I) | Wr(w*At+I)]
// softmax(rowmax-e) == exp(rowmin-e)/sum(exp(rowmin-e)).  grid = B blocks.
// ---------------------------------------------------------------------------
__global__ __launch_bounds__(256) void attn_mix_kernel(
    const float* __restrict__ energy,   // [B][2][64][64] (m=0 template)
    const float* __restrict__ gamma, const float* __restrict__ omega,
    const float* __restrict__ conv_w,   // [64][128]
    float* __restrict__ Mout) {         // [B][64][128]
  __shared__ float sAttnT[64][64];
  __shared__ float sAttnR[64][64];
  const int b = blockIdx.x;
  const int t = threadIdx.x;

  if (t < 128) {
    const int m = t >> 6, row = t & 63;
    const float* e = energy + ((size_t)(b * 2 + m)) * 4096 + row * 64;
    float mn = e[0];
#pragma unroll 8
    for (int d = 1; d < 64; ++d) mn = fminf(mn, e[d]);
    float s = 0.f;
#pragma unroll 8
    for (int d = 0; d < 64; ++d) s += __expf(mn - e[d]);
    const float inv = 1.f / s;
    float* dst = (m == 0 ? &sAttnT[0][0] : &sAttnR[0][0]) + row * 64;
#pragma unroll 8
    for (int d = 0; d < 64; ++d) dst[d] = __expf(mn - e[d]) * inv;
  }
  __syncthreads();

  const float g = gamma[0], w = omega[0];
  // M[o][c]    = g * (Wt @ attnR)[o][c] + Wt[o][c]
  // M[o][64+c] = w * (Wr @ attnT)[o][c] + Wr[o][c]
  for (int i = t; i < 64 * 128; i += 256) {
    const int o = i >> 7, c2 = i & 127;
    const int c = c2 & 63;
    const bool second = c2 >= 64;
    const float* W = conv_w + o * 128 + (second ? 64 : 0);
    const float* A = second ? &sAttnT[0][0] : &sAttnR[0][0];
    float s = 0.f;
#pragma unroll 8
    for (int d = 0; d < 64; ++d) s += W[d] * A[d * 64 + c];
    Mout[((size_t)b * 64 + o) * 128 + c2] = (second ? w : g) * s + W[c];
  }
}

// ---------------------------------------------------------------------------
// Kernel 3: out[b] = M[b](64x128) @ V[b](128x16384) + bias, V = [tmap;rmap].
// grid = B * (N/128) blocks; each block computes a 64x128 output strip.
// V strip is transposed 4x4 in registers and stored packed (ds_store_b64).
// 8 waves: wave w owns N-column-group w, 4 M-row tiles -> 16 WMMAs/wave.
// ---------------------------------------------------------------------------
__global__ __launch_bounds__(256) void apply_kernel(
    const float* __restrict__ tmap, const float* __restrict__ rmap,
    const float* __restrict__ Mmat,   // [B][64][128]
    const float* __restrict__ conv_b, // [64]
    float* __restrict__ out) {        // [B][64][16384]
  __shared__ __attribute__((aligned(16))) __bf16 sM [64][128 + PAD];
  __shared__ __attribute__((aligned(16))) __bf16 sVt[128][128 + PAD]; // [n][c]

  const int blk   = blockIdx.x;
  const int strip = blk & 127;           // N_/128 strips
  const int b     = blk >> 7;
  const int nbase = strip * 128;

  const int t    = threadIdx.x;
  const int lane = t & 31;
  const int wave = t >> 5;
  const int lm   = lane & 15;
  const bool hiH = lane >= 16;

  // ---- stage M[b] -> bf16 LDS, packed b128 stores (32 floats/thread) ----
  {
    const float* Mp = Mmat + (size_t)b * 64 * 128;
#pragma unroll
    for (int it = 0; it < 4; ++it) {
      int idx = t * 8 + it * 2048;       // row-major over [64][128]
      int ro = idx >> 7, co = idx & 127;
      float4 fa = *(const float4*)(Mp + idx);
      float4 fb = *(const float4*)(Mp + idx + 4);
      *(v8bf*)&sM[ro][co] = pack8(fa, fb);
    }
  }
  // ---- stage V strip transposed: 4x4 register transpose, b64 stores ----
#pragma unroll
  for (int it = 0; it < 4; ++it) {
    int bi = t + it * 256;               // 1024 4x4 blocks over [c:128][n:128]
    int c0 = (bi >> 5) * 4;              // 0..124
    int n0 = (bi & 31) * 4;              // 0..124
    const float* base =
        (c0 < 64 ? tmap + ((size_t)b * 64 + c0) * N_
                 : rmap + ((size_t)b * 64 + (c0 - 64)) * N_) + nbase + n0;
    float4 r0 = *(const float4*)(base);
    float4 r1 = *(const float4*)(base + N_);
    float4 r2 = *(const float4*)(base + 2 * N_);
    float4 r3 = *(const float4*)(base + 3 * N_);
    v4bf q0 = {(__bf16)r0.x, (__bf16)r1.x, (__bf16)r2.x, (__bf16)r3.x};
    v4bf q1 = {(__bf16)r0.y, (__bf16)r1.y, (__bf16)r2.y, (__bf16)r3.y};
    v4bf q2 = {(__bf16)r0.z, (__bf16)r1.z, (__bf16)r2.z, (__bf16)r3.z};
    v4bf q3 = {(__bf16)r0.w, (__bf16)r1.w, (__bf16)r2.w, (__bf16)r3.w};
    *(v4bf*)&sVt[n0 + 0][c0] = q0;
    *(v4bf*)&sVt[n0 + 1][c0] = q1;
    *(v4bf*)&sVt[n0 + 2][c0] = q2;
    *(v4bf*)&sVt[n0 + 3][c0] = q3;
  }
  __syncthreads();

  v8f acc[4] = {{}, {}, {}, {}};
#pragma unroll
  for (int kk = 0; kk < 128; kk += 32) {
    // B fragment: column n = wave*16 + lm, K contiguous along c in sVt
    const __bf16* br = &sVt[wave * 16 + lm][kk + (hiH ? 16 : 0)];
    v16bf bb = cat8(*(const v8bf*)br, *(const v8bf*)(br + 8));
#pragma unroll
    for (int ci = 0; ci < 4; ++ci) {
      const __bf16* ar = &sM[ci * 16 + lm][kk];
      v16bf a = cat8(*(const v8bf*)(ar + (hiH ? 16 : 0)),
                     *(const v8bf*)(ar + (hiH ? 24 : 8)));
      acc[ci] = wmma_bf16(a, bb, acc[ci]);
    }
  }

  // ---- epilogue: bias + store (16 consecutive floats per half-wave) ----
  const int col = nbase + wave * 16 + lm;
#pragma unroll
  for (int ci = 0; ci < 4; ++ci) {
#pragma unroll
    for (int v = 0; v < 8; ++v) {
      const int o = ci * 16 + v + (hiH ? 8 : 0);
      out[((size_t)b * 64 + o) * N_ + col] = acc[ci][v] + conv_b[o];
    }
  }
}

// ---------------------------------------------------------------------------
extern "C" void kernel_launch(void* const* d_in, const int* in_sizes, int n_in,
                              void* d_out, int out_size, void* d_ws, size_t ws_size,
                              hipStream_t stream) {
  const float* tmap   = (const float*)d_in[0];
  const float* rmap   = (const float*)d_in[1];
  const float* gamma  = (const float*)d_in[2];
  const float* omega  = (const float*)d_in[3];
  const float* conv_w = (const float*)d_in[4];
  const float* conv_b = (const float*)d_in[5];
  float* out = (float*)d_out;

  float* energy = (float*)d_ws;                      // [16][2][64][64] = 512 KB
  float* Mmat   = energy + (size_t)B_ * 2 * 64 * 64; // [16][64][128]   = 512 KB

  hipMemsetAsync(energy, 0, (size_t)B_ * 2 * 64 * 64 * sizeof(float), stream);

  energy_kernel<<<dim3(B_ * 2 * KSPLIT), 256, 0, stream>>>(tmap, rmap, energy);
  attn_mix_kernel<<<dim3(B_), 256, 0, stream>>>(energy, gamma, omega, conv_w, Mmat);
  apply_kernel<<<dim3(B_ * 128), 256, 0, stream>>>(tmap, rmap, Mmat, conv_b, out);
}
